// MultiHeadAttention_30142080483478
// MI455X (gfx1250) — matrix operand order repaired and checked by reading, not compiled, not run on hardware
//
#include <hip/hip_runtime.h>
#include <hip/hip_bf16.h>

typedef __attribute__((ext_vector_type(16))) _Float16 v16h;
typedef __attribute__((ext_vector_type(8)))  _Float16 v8h;
typedef __attribute__((ext_vector_type(8)))  float    v8f;

#define BN     64      // B * N_NEWS
#define T_SEQ  512
#define D_EMB  256
#define NH     8
#define DQ     32
#define NEGBIG (-1e9f)

// ---- WMMA fragment helpers (wave32, 16x16x32 f16 -> f32) --------------------

// A fragment: 16 rows (M) x 32 (K), row-major source, `stride` halfs between rows.
// Lane L (L<16): row M=L, K = {0..7} then {16..23}; lane L+16: K = {8..15},{24..31}.
__device__ __forceinline__ v16h load_a_frag(const _Float16* tile, int stride) {
    int lane = threadIdx.x & 31;
    int r = lane & 15;
    int hi = lane >> 4;
    const _Float16* p = tile + r * stride + hi * 8;
    v8h lo = *(const v8h*)(p);
    v8h hv = *(const v8h*)(p + 16);
    v16h f;
#pragma unroll
    for (int i = 0; i < 8; ++i) { f[i] = lo[i]; f[i + 8] = hv[i]; }
    return f;
}

// B fragment: 32 (K) x 16 cols (N); columns stored contiguously (`stride` halfs
// between columns). Lane L (L<16): col N=L, K = {0..7},{16..23}; lane L+16: K = {8..15},{24..31}.
__device__ __forceinline__ v16h load_b_frag(const _Float16* tile, int stride) {
    int lane = threadIdx.x & 31;
    int c = lane & 15;
    int hi = lane >> 4;
    const _Float16* p = tile + c * stride + hi * 8;
    v8h lo = *(const v8h*)(p);
    v8h hv = *(const v8h*)(p + 16);
    v16h f;
#pragma unroll
    for (int i = 0; i < 8; ++i) { f[i] = lo[i]; f[i + 8] = hv[i]; }
    return f;
}

__device__ __forceinline__ v8f wmma_f16(v16h a, v16h b, v8f c) {
    return __builtin_amdgcn_wmma_f32_16x16x32_f16(false, a, false, b, (short)0, c,
                                                  false, false);
}

// ---- Conversion kernels -----------------------------------------------------

__global__ void conv_x_kernel(const float* __restrict__ x, _Float16* __restrict__ xh, int n) {
    int i = blockIdx.x * blockDim.x + threadIdx.x;
    if (i < n) xh[i] = (_Float16)x[i];
}

// WT[m][o][d] = W_m[h][d][j] with o = h*32+j  (columns of B contiguous per output o)
__global__ void conv_w_kernel(const float* __restrict__ Wq, const float* __restrict__ Wk,
                              const float* __restrict__ Wv, _Float16* __restrict__ wt) {
    int i = blockIdx.x * blockDim.x + threadIdx.x;   // < 3*256*256
    int m = i >> 16;
    int o = (i >> 8) & 255;
    int d = i & 255;
    const float* W = (m == 0) ? Wq : (m == 1 ? Wk : Wv);
    int h = o >> 5, j = o & 31;
    wt[i] = (_Float16)W[(h * D_EMB + d) * DQ + j];
}

// y = a @ Wp.T  => B column j of the GEMM is row j of Wp: direct f16 copy.
__global__ void conv_wp_kernel(const float* __restrict__ Wp, _Float16* __restrict__ wpt, int n) {
    int i = blockIdx.x * blockDim.x + threadIdx.x;
    if (i < n) wpt[i] = (_Float16)Wp[i];
}

// ---- QKV projection: per (bn), (512x256)@(256x256) for Wq/Wk/Wv -------------
// grid = (4, 64), block = 256 (8 waves); each wave owns a 16-row tile.

__global__ void qkv_kernel(const _Float16* __restrict__ xh, const _Float16* __restrict__ wt,
                           _Float16* __restrict__ qh, _Float16* __restrict__ kh,
                           _Float16* __restrict__ vt) {
    int bn = blockIdx.y;
    int wave = threadIdx.x >> 5;
    int lane = threadIdx.x & 31;
    int t0 = blockIdx.x * 128 + wave * 16;
    const _Float16* arow = xh + (size_t)(bn * T_SEQ + t0) * D_EMB;

    v16h afrag[8];
#pragma unroll
    for (int kt = 0; kt < 8; ++kt) afrag[kt] = load_a_frag(arow + kt * 32, D_EMB);

    int c = lane & 15, hi = lane >> 4;
#pragma unroll
    for (int m = 0; m < 3; ++m) {
        const _Float16* wbase = wt + m * 65536;
        for (int nt = 0; nt < 16; ++nt) {
            v8f acc = {};
#pragma unroll
            for (int kt = 0; kt < 8; ++kt) {
                v16h b = load_b_frag(wbase + (nt * 16) * D_EMB + kt * 32, D_EMB);
                acc = wmma_f16(afrag[kt], b, acc);
            }
            int o = nt * 16 + c;
            int h = o >> 5, j = o & 31;
            if (m < 2) {
                // Q,K: [bn][h][t][j] row-major per head
                _Float16* dst = (m == 0 ? qh : kh) + ((size_t)(bn * NH + h) * T_SEQ) * DQ + j;
#pragma unroll
                for (int rr = 0; rr < 8; ++rr) {
                    int t = t0 + rr + hi * 8;
                    dst[t * DQ] = (_Float16)acc[rr];
                }
            } else {
                // V transposed: [bn][h][j][t]  (columns of V contiguous)
                _Float16* dst = vt + ((size_t)(bn * NH + h) * DQ + j) * T_SEQ;
#pragma unroll
                for (int rr = 0; rr < 8; ++rr) {
                    int t = t0 + rr + hi * 8;
                    dst[t] = (_Float16)acc[rr];
                }
            }
        }
    }
}

// ---- Attention: one wave per (bn, head, 16-query tile) ----------------------
// grid = (32, 8, 64), block = 32.

__global__ void attn_kernel(const _Float16* __restrict__ qh, const _Float16* __restrict__ kh,
                            const _Float16* __restrict__ vt,
                            const unsigned char* __restrict__ mask,
                            _Float16* __restrict__ ao) {
    __shared__ __align__(16) float    sL[16 * T_SEQ];   // 32 KB logits
    __shared__ __align__(16) _Float16 sP[16 * T_SEQ];   // 16 KB exp(P)
    __shared__ float sInv[16];

    int qt = blockIdx.x;
    int h  = blockIdx.y;
    int bn = blockIdx.z;
    int lane = threadIdx.x;
    int c = lane & 15, hi = lane >> 4;
    int t0 = qt * 16;

    const _Float16* qhead = qh + ((size_t)(bn * NH + h) * T_SEQ) * DQ;
    const _Float16* khead = kh + ((size_t)(bn * NH + h) * T_SEQ) * DQ;
    const _Float16* vhead = vt + ((size_t)(bn * NH + h) * DQ) * T_SEQ;
    const unsigned char* mrow = mask + (size_t)bn * T_SEQ;

    v16h aq = load_a_frag(qhead + t0 * DQ, DQ);

    // logits strip (16 x 512), mask folded in at store time
    for (int st = 0; st < 32; ++st) {
        v16h bk = load_b_frag(khead + (st * 16) * DQ, DQ);
        v8f acc = {};
        acc = wmma_f16(aq, bk, acc);
        int s = st * 16 + c;
        bool keep = mrow[s] != 0;
#pragma unroll
        for (int rr = 0; rr < 8; ++rr)
            sL[(rr + hi * 8) * T_SEQ + s] = keep ? acc[rr] : NEGBIG;
    }
    __syncthreads();

    // row softmax: max-subtracted exp into sP (f16, unnormalized), 1/sum kept
    for (int r = 0; r < 16; ++r) {
        float mx = NEGBIG;
        for (int i = 0; i < 16; ++i)
            mx = fmaxf(mx, sL[r * T_SEQ + lane + i * 32]);
#pragma unroll
        for (int off = 16; off > 0; off >>= 1) mx = fmaxf(mx, __shfl_xor(mx, off, 32));
        float sum = 0.f;
        for (int i = 0; i < 16; ++i) {
            int cc = lane + i * 32;
            float p = __expf(sL[r * T_SEQ + cc] - mx);
            sP[r * T_SEQ + cc] = (_Float16)p;
            sum += p;
        }
#pragma unroll
        for (int off = 16; off > 0; off >>= 1) sum += __shfl_xor(sum, off, 32);
        if (lane == 0) sInv[r] = 1.0f / sum;
    }
    __syncthreads();

    // out = P @ V : K = 512 (16 k-tiles), N = 32 (2 n-tiles); normalize on store
    _Float16* aorow = ao + (size_t)(bn * T_SEQ + t0) * (NH * DQ) + h * DQ;
#pragma unroll
    for (int j0 = 0; j0 < DQ; j0 += 16) {
        v8f acc = {};
        for (int kt = 0; kt < 16; ++kt) {
            v16h ap = load_a_frag((const _Float16*)&sP[kt * 32], T_SEQ);   // ds_load path
            v16h bv = load_b_frag(vhead + j0 * T_SEQ + kt * 32, T_SEQ);
            acc = wmma_f16(ap, bv, acc);
        }
#pragma unroll
        for (int rr = 0; rr < 8; ++rr) {
            int m = rr + hi * 8;
            aorow[m * (NH * DQ) + j0 + c] = (_Float16)(acc[rr] * sInv[m]);
        }
    }
}

// ---- Output projection: (32768x256)@(256x256) + bias, f32 out ---------------
// grid = 256, block = 256 (8 waves x 16 rows)

__global__ void proj_kernel(const _Float16* __restrict__ ao, const _Float16* __restrict__ wpt,
                            const float* __restrict__ bp, float* __restrict__ out) {
    int wave = threadIdx.x >> 5;
    int lane = threadIdx.x & 31;
    int g = blockIdx.x * 128 + wave * 16;
    const _Float16* arow = ao + (size_t)g * D_EMB;

    v16h afrag[8];
#pragma unroll
    for (int kt = 0; kt < 8; ++kt) afrag[kt] = load_a_frag(arow + kt * 32, D_EMB);

    int c = lane & 15, hi = lane >> 4;
    for (int nt = 0; nt < 16; ++nt) {
        v8f acc = {};
#pragma unroll
        for (int kt = 0; kt < 8; ++kt) {
            v16h b = load_b_frag(wpt + (nt * 16) * D_EMB + kt * 32, D_EMB);
            acc = wmma_f16(afrag[kt], b, acc);
        }
        int col = nt * 16 + c;
        float bias = bp[col];
#pragma unroll
        for (int rr = 0; rr < 8; ++rr)
            out[(size_t)(g + rr + hi * 8) * D_EMB + col] = acc[rr] + bias;
    }
}

// ---- Host launch ------------------------------------------------------------

extern "C" void kernel_launch(void* const* d_in, const int* in_sizes, int n_in,
                              void* d_out, int out_size, void* d_ws, size_t ws_size,
                              hipStream_t stream) {
    const float* x            = (const float*)d_in[0];
    const unsigned char* mask = (const unsigned char*)d_in[1];  // bool array, 1 byte/elem
    const float* Wq           = (const float*)d_in[2];
    const float* Wk           = (const float*)d_in[3];
    const float* Wv           = (const float*)d_in[4];
    const float* Wp           = (const float*)d_in[5];
    const float* bp           = (const float*)d_in[6];
    float* out                = (float*)d_out;

    _Float16* ws = (_Float16*)d_ws;
    const size_t SZX = (size_t)BN * T_SEQ * D_EMB;      // 8,388,608 halfs
    _Float16* XH  = ws;                                 // x in f16
    _Float16* WT  = XH + SZX;                           // 3 x 256 x 256 (pre-transposed)
    _Float16* WPT = WT + 3 * 65536;                     // 256 x 256
    _Float16* QH  = WPT + 65536;                        // [bn][h][t][dq]
    _Float16* KH  = QH + SZX;                           // [bn][h][t][dq]
    _Float16* VT  = KH + SZX;                           // [bn][h][dq][t]
    _Float16* AO  = VT + SZX;                           // [bn][t][h*dq]

    conv_x_kernel<<<(unsigned)(SZX / 256), 256, 0, stream>>>(x, XH, (int)SZX);
    conv_w_kernel<<<(3 * 65536) / 256, 256, 0, stream>>>(Wq, Wk, Wv, WT);
    conv_wp_kernel<<<65536 / 256, 256, 0, stream>>>(Wp, WPT, 65536);
    qkv_kernel<<<dim3(4, BN), 256, 0, stream>>>(XH, WT, QH, KH, VT);
    attn_kernel<<<dim3(32, NH, BN), 32, 0, stream>>>(QH, KH, VT, mask, AO);
    proj_kernel<<<256, 256, 0, stream>>>(AO, WPT, bp, out);
}
